// GAT_34273839022828
// MI455X (gfx1250) — compile-verified, matrix operand order checked by reading, not run on hardware
//
#include <hip/hip_runtime.h>

#define N_NODES 10000
#define N_EDGES 320000
#define DIM     256
#define NEG_SLOPE 0.2f

typedef __attribute__((ext_vector_type(16))) __bf16 v16bf;
typedef __attribute__((ext_vector_type(8)))  float  v8f;

// ---------- helpers ----------

__device__ __forceinline__ unsigned short f2bf(float f) {
    unsigned u = __float_as_uint(f);
    unsigned r = u + 0x7FFFu + ((u >> 16) & 1u);   // round-to-nearest-even
    return (unsigned short)(r >> 16);
}

// order-preserving float<->uint encoding for atomicMax on signed floats
__device__ __forceinline__ unsigned enc_f32(float f) {
    unsigned u = __float_as_uint(f);
    return (u & 0x80000000u) ? ~u : (u | 0x80000000u);
}
__device__ __forceinline__ float dec_f32(unsigned u) {
    unsigned v = (u & 0x80000000u) ? (u ^ 0x80000000u) : ~u;
    return __uint_as_float(v);
}

// K-index inside a 32-wide K tile for the 16-bit WMMA A/B fragment layout:
// lanes 0-15 hold K blocks {0..7,16..23}, lanes 16-31 hold {8..15,24..31};
// VGPR v packs K pair (2v, 2v+1) within each half.
__device__ __forceinline__ int frag_k(int vg, int sub, int half) {
    return (vg < 4) ? (vg * 2 + sub + half * 8)
                    : (16 + (vg - 4) * 2 + sub + half * 8);
}

// ---------- kernel 1: pack W (f32 -> bf16, B-fragment swizzled layout) ----------
// Wp element index = (((ktile*16 + ntile)*32 + lane)*16 + i)
__global__ void gat_pack_W(const float* __restrict__ W, unsigned short* __restrict__ Wp) {
    int tid   = blockIdx.x * 256 + threadIdx.x;     // 65536 total
    int i     = tid & 15;
    int lane  = (tid >> 4) & 31;
    int ntile = (tid >> 9) & 15;
    int ktile = tid >> 13;

    int half = lane >> 4, col = lane & 15;
    int vg = i >> 1, sub = i & 1;
    int K = ktile * 32 + frag_k(vg, sub, half);
    int n = ntile * 16 + col;
    Wp[tid] = f2bf(W[K * DIM + n]);
}

// ---------- kernel 2: h = feats @ W via v_wmma_f32_16x16x32_bf16 ----------
// grid 625 blocks (one 16-row M tile each), 256 threads = 8 waves,
// each wave owns two 16-col N tiles.
__global__ void __launch_bounds__(256)
gat_gemm(const float* __restrict__ feats, const unsigned short* __restrict__ Wp,
         float* __restrict__ h) {
    __shared__ unsigned short Atile[16][DIM + 8];   // bf16, padded stride (bank-conflict free)

    const int t     = threadIdx.x;
    const int mtile = blockIdx.x;

    // cooperative A-tile load: 16 rows x 256 cols f32 -> bf16 LDS
    {
        int row = t >> 4;
        int cb  = (t & 15) * 16;
        const float4* fp = (const float4*)(feats + (mtile * 16 + row) * DIM + cb);
        #pragma unroll
        for (int j = 0; j < 4; ++j) {
            float4 v = fp[j];
            Atile[row][cb + j * 4 + 0] = f2bf(v.x);
            Atile[row][cb + j * 4 + 1] = f2bf(v.y);
            Atile[row][cb + j * 4 + 2] = f2bf(v.z);
            Atile[row][cb + j * 4 + 3] = f2bf(v.w);
        }
    }
    __syncthreads();

    const int wave = t >> 5;
    const int lane = t & 31;
    const int row  = lane & 15;
    const int half = lane >> 4;
    const int nt0  = wave * 2;
    const int nt1  = wave * 2 + 1;

    union FragA { v16bf v; unsigned u[8]; };
    union FragB { v16bf v; uint4 q[2]; };

    v8f c0 = {}; v8f c1 = {};

    #pragma unroll
    for (int ktile = 0; ktile < 8; ++ktile) {
        FragA a;
        #pragma unroll
        for (int vg = 0; vg < 8; ++vg) {
            int k0 = frag_k(vg, 0, half);           // k0,k0+1 contiguous
            a.u[vg] = *(const unsigned*)&Atile[row][ktile * 32 + k0];
        }
        FragB b0, b1;
        {
            const uint4* p0 = (const uint4*)(Wp + (((ktile * 16 + nt0) * 32 + lane) << 4));
            b0.q[0] = p0[0]; b0.q[1] = p0[1];
            const uint4* p1 = (const uint4*)(Wp + (((ktile * 16 + nt1) * 32 + lane) << 4));
            b1.q[0] = p1[0]; b1.q[1] = p1[1];
        }
        c0 = __builtin_amdgcn_wmma_f32_16x16x32_bf16(false, a.v, false, b0.v,
                                                     (short)0, c0, false, false);
        c1 = __builtin_amdgcn_wmma_f32_16x16x32_bf16(false, a.v, false, b1.v,
                                                     (short)0, c1, false, false);
    }

    // D layout: VGPR r -> M = r (lanes 0-15) / r+8 (lanes 16-31), N = lane%16
    const int mrow = mtile * 16 + ((half == 0) ? 0 : 8);
    const int nc   = lane & 15;
    #pragma unroll
    for (int r = 0; r < 8; ++r) {
        h[(mrow + r) * DIM + nt0 * 16 + nc] = c0[r];
        h[(mrow + r) * DIM + nt1 * 16 + nc] = c1[r];
    }
}

// ---------- kernel 3: per-node el/er dots + init m,s,out ----------
__global__ void __launch_bounds__(256)
gat_node_prep(const float* __restrict__ h, const float* __restrict__ attn_l,
              const float* __restrict__ attn_r, float* __restrict__ el,
              float* __restrict__ er, unsigned* __restrict__ m_enc,
              float* __restrict__ s_sum, float* __restrict__ out) {
    const int wave = threadIdx.x >> 5;
    const int lane = threadIdx.x & 31;
    const int node = blockIdx.x * 8 + wave;

    const int cb = lane * 8;
    const float4* hp = (const float4*)(h + node * DIM + cb);
    const float4* lp = (const float4*)(attn_l + cb);
    const float4* rp = (const float4*)(attn_r + cb);
    float4 h0 = hp[0], h1 = hp[1];
    float4 l0 = lp[0], l1 = lp[1];
    float4 r0 = rp[0], r1 = rp[1];

    float dl = h0.x * l0.x + h0.y * l0.y + h0.z * l0.z + h0.w * l0.w
             + h1.x * l1.x + h1.y * l1.y + h1.z * l1.z + h1.w * l1.w;
    float dr = h0.x * r0.x + h0.y * r0.y + h0.z * r0.z + h0.w * r0.w
             + h1.x * r1.x + h1.y * r1.y + h1.z * r1.z + h1.w * r1.w;
    #pragma unroll
    for (int off = 16; off > 0; off >>= 1) {
        dl += __shfl_xor(dl, off, 32);
        dr += __shfl_xor(dr, off, 32);
    }
    if (lane == 0) {
        el[node] = dl;
        er[node] = dr;
        m_enc[node] = enc_f32(-__builtin_inff());
        s_sum[node] = 0.0f;
    }
    // zero the output accumulator (harness poisons d_out)
    float4* op = (float4*)(out + node * DIM + cb);
    op[0] = make_float4(0.f, 0.f, 0.f, 0.f);
    op[1] = make_float4(0.f, 0.f, 0.f, 0.f);
}

// ---------- kernel 4: edge leaky-relu + segment max ----------
__global__ void __launch_bounds__(256)
gat_edge_max(const int* __restrict__ src, const int* __restrict__ dst,
             const float* __restrict__ el, const float* __restrict__ er,
             float* __restrict__ e_buf, unsigned* __restrict__ m_enc) {
    int i = blockIdx.x * 256 + threadIdx.x;         // exactly N_EDGES threads
    int s = src[i], d = dst[i];
    float e = el[s] + er[d];
    e = (e > 0.f) ? e : NEG_SLOPE * e;
    e_buf[i] = e;
    atomicMax(&m_enc[d], enc_f32(e));
}

// ---------- kernel 5: wave-per-edge exp + denominator + weighted scatter ----------
__global__ void __launch_bounds__(256)
gat_edge_agg(const int* __restrict__ src, const int* __restrict__ dst,
             const float* __restrict__ e_buf, const unsigned* __restrict__ m_enc,
             float* __restrict__ s_sum, const float* __restrict__ h,
             float* __restrict__ out) {
    const int wave = threadIdx.x >> 5;
    const int lane = threadIdx.x & 31;
    const int edge = blockIdx.x * 8 + wave;         // 40000 blocks * 8 = N_EDGES

    const int sN = src[edge];
    const int d  = dst[edge];
    const float ex = __expf(e_buf[edge] - dec_f32(m_enc[d]));
    if (lane == 0) atomicAdd(&s_sum[d], ex);

    const int cb = lane * 8;
    const float4* hp = (const float4*)(h + sN * DIM + cb);
    float4 h0 = hp[0], h1 = hp[1];
    float* op = out + d * DIM + cb;
    atomicAdd(op + 0, ex * h0.x);
    atomicAdd(op + 1, ex * h0.y);
    atomicAdd(op + 2, ex * h0.z);
    atomicAdd(op + 3, ex * h0.w);
    atomicAdd(op + 4, ex * h1.x);
    atomicAdd(op + 5, ex * h1.y);
    atomicAdd(op + 6, ex * h1.z);
    atomicAdd(op + 7, ex * h1.w);
}

// ---------- kernel 6: normalize + bias ----------
__global__ void __launch_bounds__(256)
gat_finalize(float* __restrict__ out, const float* __restrict__ s_sum,
             const float* __restrict__ bias) {
    const int wave = threadIdx.x >> 5;
    const int lane = threadIdx.x & 31;
    const int node = blockIdx.x * 8 + wave;
    const float sv = s_sum[node];
    const float inv = (sv > 0.f) ? (1.0f / sv) : 0.0f;   // empty segment -> bias only

    const int cb = lane * 8;
    float4* op = (float4*)(out + node * DIM + cb);
    const float4* bp = (const float4*)(bias + cb);
    float4 o0 = op[0], o1 = op[1];
    float4 b0 = bp[0], b1 = bp[1];
    o0.x = o0.x * inv + b0.x; o0.y = o0.y * inv + b0.y;
    o0.z = o0.z * inv + b0.z; o0.w = o0.w * inv + b0.w;
    o1.x = o1.x * inv + b1.x; o1.y = o1.y * inv + b1.y;
    o1.z = o1.z * inv + b1.z; o1.w = o1.w * inv + b1.w;
    op[0] = o0; op[1] = o1;
}

// ---------- launch ----------
extern "C" void kernel_launch(void* const* d_in, const int* in_sizes, int n_in,
                              void* d_out, int out_size, void* d_ws, size_t ws_size,
                              hipStream_t stream) {
    const float* feats  = (const float*)d_in[0];
    const float* W      = (const float*)d_in[1];
    const float* attn_l = (const float*)d_in[2];
    const float* attn_r = (const float*)d_in[3];
    const float* bias   = (const float*)d_in[4];
    const int*   src    = (const int*)d_in[5];
    const int*   dst    = (const int*)d_in[6];
    float* out = (float*)d_out;

    // workspace layout (bytes)
    char* ws = (char*)d_ws;
    float*          h     = (float*)(ws);                               // 10,240,000
    unsigned short* Wp    = (unsigned short*)(ws + 10240000);           //    131,072
    float*          el    = (float*)(ws + 10371072);                    //     40,000
    float*          er    = (float*)(ws + 10411072);                    //     40,000
    unsigned*       m_enc = (unsigned*)(ws + 10451072);                 //     40,000
    float*          s_sum = (float*)(ws + 10491072);                    //     40,000
    float*          e_buf = (float*)(ws + 10531072);                    //  1,280,000

    gat_pack_W   <<<DIM * DIM / 256, 256, 0, stream>>>(W, Wp);
    gat_gemm     <<<N_NODES / 16,    256, 0, stream>>>(feats, Wp, h);
    gat_node_prep<<<N_NODES / 8,     256, 0, stream>>>(h, attn_l, attn_r, el, er,
                                                       m_enc, s_sum, out);
    gat_edge_max <<<N_EDGES / 256,   256, 0, stream>>>(src, dst, el, er, e_buf, m_enc);
    gat_edge_agg <<<N_EDGES / 8,     256, 0, stream>>>(src, dst, e_buf, m_enc,
                                                       s_sum, h, out);
    gat_finalize <<<N_NODES / 8,     256, 0, stream>>>(out, s_sum, bias);
}
